// PolyTensorRegression_14293651161652
// MI455X (gfx1250) — matrix operand order repaired and checked by reading, not compiled, not run on hardware
//
#include <hip/hip_runtime.h>
#include <hip/hip_bf16.h>
#include <stdint.h>

// Problem constants (from reference)
#define B_ROWS   16384
#define D_DIM    2048
#define RANK     128
#define NFAC     10
#define KPAIRS   (D_DIM / 2)          // 1024 packed bf16 k-pairs per (f, r) row
#define REG_W_C  0.1f
#define REG_C_C  0.05f

typedef __attribute__((ext_vector_type(16))) __bf16 v16bf;
typedef __attribute__((ext_vector_type(8)))  float  v8f;

union BFrag { v16bf v; uint4 q[2]; unsigned int u[8]; };
union AccU  { v8f   v; float f[8]; };

// round-to-nearest-even f32 -> bf16, pack two into one u32 (lo in [15:0])
__device__ __forceinline__ unsigned int pack_bf16(float lo, float hi) {
    unsigned int a = __float_as_uint(lo), b = __float_as_uint(hi);
    a += 0x7FFFu + ((a >> 16) & 1u);
    b += 0x7FFFu + ((b >> 16) & 1u);
    return (a >> 16) | (b & 0xFFFF0000u);
}

// ---------------------------------------------------------------------------
// Pre-pass: repack U_all [10][2048][128] f32 into B-fragment-friendly layout
// Upk[f][r][kp] = pack( bf16(U[f][2kp][r]), bf16(U[f][2kp+1][r]) )
// so a wave's B fragment (lane L, vgprs v=0..7: K=(L/16)*16+2v) is 8
// consecutive u32 -> two global_load_b128, zero conversion VALU in hot loop.
// ---------------------------------------------------------------------------
__global__ void convert_U_kernel(const float* __restrict__ U,
                                 unsigned int* __restrict__ Upk) {
    int o = blockIdx.x * blockDim.x + threadIdx.x;
    if (o >= NFAC * RANK * KPAIRS) return;
    int f   = o / (RANK * KPAIRS);
    int rem = o - f * (RANK * KPAIRS);
    int r   = rem / KPAIRS;
    int kp  = rem - r * KPAIRS;
    const float* base = U + (size_t)f * D_DIM * RANK + r;
    float lo = base[(size_t)(2 * kp)     * RANK];
    float hi = base[(size_t)(2 * kp + 1) * RANK];
    Upk[o] = pack_bf16(lo, hi);
}

// ---------------------------------------------------------------------------
// Main fused kernel. Grid: 512 WGs x 256 threads (8 waves, wave32).
// WG covers 32 rows of X. Wave w owns r-tile w (16 rank columns) and ALL 10
// factor matrices for 2 M-tiles -> 20 f32 16x16 accumulators per wave.
// B fragments are staged 5-deep in registers; sched_barrier keeps the load
// cluster ahead of the WMMA cluster so up to 10 b128 loads stay in flight
// underneath the matrix ops.
// Epilogue (CP products across factors) is elementwise in registers, then a
// deterministic LDS tree reduction over r.
// ---------------------------------------------------------------------------
__global__ __launch_bounds__(256, 1)
void poly_gemm_kernel(const float* __restrict__ X,
                      const unsigned int* __restrict__ Upk,
                      const float* __restrict__ C,
                      const float* __restrict__ betap,
                      float* __restrict__ out) {
    __shared__ unsigned int sA[32][20];   // 32 rows x 16 packed k-pairs (+pad: 20*m mod 64 distinct)
    __shared__ float red[32][132];        // per-(row, r) combined value (+pad)
    __shared__ float red2[32][8];

    const int tid  = threadIdx.x;
    const int lane = tid & 31;
    const int wave = tid >> 5;            // r-tile id, 0..7
    const int l16  = lane & 15;
    const int lhi  = lane >> 4;           // 0 or 1
    const int row0 = blockIdx.x * 32;
    const int r    = wave * 16 + l16;     // this lane's rank column

    AccU acc[2][NFAC];
    const v8f zero = {0.f, 0.f, 0.f, 0.f, 0.f, 0.f, 0.f, 0.f};
    #pragma unroll
    for (int mt = 0; mt < 2; ++mt)
        #pragma unroll
        for (int f = 0; f < NFAC; ++f)
            acc[mt][f].v = zero;

    // base of this lane's B-fragment column (factor stride = RANK*KPAIRS u32)
    const unsigned int* ubase = Upk + (size_t)r * KPAIRS + lhi * 8;

    for (int kc = 0; kc < D_DIM; kc += 32) {
        // ---- stage X[row0..row0+31][kc..kc+31] into LDS as packed bf16 pairs
        #pragma unroll
        for (int it = tid; it < 512; it += 256) {
            int m = it >> 4, kp = it & 15;
            const float* xp = X + (size_t)(row0 + m) * D_DIM + kc + 2 * kp;
            __builtin_prefetch(xp + 32, 0, 3);              // next K-chunk of X
            const float2 xv = *(const float2*)xp;
            sA[m][kp] = pack_bf16(xv.x, xv.y);
        }
        __syncthreads();

        const unsigned int* bp = ubase + (kc >> 1);

        // ---- A fragments: contiguous k layout (matches B) -> vgpr v: kpair=lhi*8+v
        BFrag a0, a1;
        {
            const int b0 = lhi * 8;                         // 16B aligned (row stride 80B)
            a0.q[0] = *(const uint4*)&sA[l16][b0];          // v=0..3
            a0.q[1] = *(const uint4*)&sA[l16][b0 + 4];      // v=4..7
            a1.q[0] = *(const uint4*)&sA[16 + l16][b0];
            a1.q[1] = *(const uint4*)&sA[16 + l16][b0 + 4];
        }

        // ---- 10 factors in two half-rounds; 5 B fragments staged per round.
        #pragma unroll
        for (int h = 0; h < 2; ++h) {
            BFrag bb[5];
            #pragma unroll
            for (int j = 0; j < 5; ++j) {
                const unsigned int* bpj =
                    bp + (size_t)(h * 5 + j) * (RANK * KPAIRS);
                __builtin_prefetch(bpj + 16, 0, 3);         // next K-chunk of U
                bb[j].q[0] = *(const uint4*)bpj;
                bb[j].q[1] = *(const uint4*)(bpj + 4);
            }
            // keep the 10 loads clustered ahead of the WMMAs
            __builtin_amdgcn_sched_barrier(0);
            #pragma unroll
            for (int j = 0; j < 5; ++j) {
                const int f = h * 5 + j;
                acc[0][f].v = __builtin_amdgcn_wmma_f32_16x16x32_bf16(
                    false, a0.v, false, bb[j].v, (short)0, acc[0][f].v, false, false);
                acc[1][f].v = __builtin_amdgcn_wmma_f32_16x16x32_bf16(
                    false, a1.v, false, bb[j].v, (short)0, acc[1][f].v, false, false);
            }
        }
        __syncthreads();
    }

    // ---- epilogue: CP products across factors are elementwise in registers
    const float c0 = C[0], c1 = C[1], c2 = C[2], c3 = C[3];
    #pragma unroll
    for (int mt = 0; mt < 2; ++mt) {
        #pragma unroll
        for (int i = 0; i < 8; ++i) {
            float d0 = acc[mt][0].f[i], d1 = acc[mt][1].f[i], d2 = acc[mt][2].f[i];
            float d3 = acc[mt][3].f[i], d4 = acc[mt][4].f[i], d5 = acc[mt][5].f[i];
            float d6 = acc[mt][6].f[i], d7 = acc[mt][7].f[i], d8 = acc[mt][8].f[i];
            float d9 = acc[mt][9].f[i];
            float val = c0 * d0
                      + c1 * (d1 * d2)
                      + c2 * (d3 * d4 * d5)
                      + c3 * ((d6 * d7) * (d8 * d9));
            // C/D layout: vgpr i -> M=i (lanes 0-15) or M=i+8 (lanes 16-31)
            int m = mt * 16 + i + (lhi << 3);
            red[m][r] = val;
        }
    }
    __syncthreads();

    // deterministic tree reduction over r (128 values per row)
    {
        int rowr = tid >> 3, seg = tid & 7;
        float s = 0.f;
        #pragma unroll
        for (int j = 0; j < 16; ++j) s += red[rowr][seg * 16 + j];
        red2[rowr][seg] = s;
    }
    __syncthreads();
    if (tid < 32) {
        float s = 0.f;
        #pragma unroll
        for (int j = 0; j < 8; ++j) s += red2[tid][j];
        out[row0 + tid] = betap[0] + s;
    }
}

// ---------------------------------------------------------------------------
// Regularizer: deterministic two-pass reduction (no fp atomics).
// element in factor f weighs REG_W * |u| / (m_f * D * R), m_f in {1,2,3,4}.
// ---------------------------------------------------------------------------
__global__ void reg_partial_kernel(const float* __restrict__ U,
                                   float* __restrict__ part) {
    __shared__ float sm[256];
    const int N = NFAC * D_DIM * RANK;
    const float inv = REG_W_C / (float)(D_DIM * RANK);
    float s = 0.f;
    for (int i = blockIdx.x * blockDim.x + threadIdx.x; i < N;
         i += gridDim.x * blockDim.x) {
        int f = i / (D_DIM * RANK);
        float w = (f == 0) ? 1.0f : (f < 3) ? 0.5f : (f < 6) ? (1.0f / 3.0f) : 0.25f;
        s += fabsf(U[i]) * w;
    }
    sm[threadIdx.x] = s * inv;
    __syncthreads();
    for (int st = 128; st > 0; st >>= 1) {
        if (threadIdx.x < st) sm[threadIdx.x] += sm[threadIdx.x + st];
        __syncthreads();
    }
    if (threadIdx.x == 0) part[blockIdx.x] = sm[0];
}

__global__ void reg_final_kernel(const float* __restrict__ part,
                                 const float* __restrict__ C,
                                 float* __restrict__ out_reg) {
    __shared__ float sm[256];
    sm[threadIdx.x] = part[threadIdx.x];
    __syncthreads();
    for (int st = 128; st > 0; st >>= 1) {
        if (threadIdx.x < st) sm[threadIdx.x] += sm[threadIdx.x + st];
        __syncthreads();
    }
    if (threadIdx.x == 0)
        out_reg[0] = sm[0] + REG_C_C * (fabsf(C[0]) + fabsf(C[1]) +
                                        fabsf(C[2]) + fabsf(C[3]));
}

// ---------------------------------------------------------------------------
// d_in: [0]=X [16384*2048] f32, [1]=U_all [10*2048*128] f32,
//       [2]=C [4] f32, [3]=beta [1] f32
// d_out: 16384 result + 1 total_reg (f32)
// d_ws:  Upk (5,242,880 B) | reg partials (256 f32)
// ---------------------------------------------------------------------------
extern "C" void kernel_launch(void* const* d_in, const int* in_sizes, int n_in,
                              void* d_out, int out_size, void* d_ws, size_t ws_size,
                              hipStream_t stream) {
    const float* X    = (const float*)d_in[0];
    const float* Uall = (const float*)d_in[1];
    const float* C    = (const float*)d_in[2];
    const float* beta = (const float*)d_in[3];
    float* out = (float*)d_out;

    unsigned int* Upk  = (unsigned int*)d_ws;
    float*        part = (float*)((char*)d_ws + (size_t)NFAC * RANK * KPAIRS * 4);

    // 1) repack U -> bf16-pair fragment layout (L2-resident, 5.2 MB)
    {
        int n = NFAC * RANK * KPAIRS;               // 1,310,720
        convert_U_kernel<<<(n + 255) / 256, 256, 0, stream>>>(Uall, Upk);
    }
    // 2) regularizer partial sums (independent of 1/3)
    reg_partial_kernel<<<256, 256, 0, stream>>>(Uall, part);
    // 3) fused WMMA GEMM + CP-product epilogue
    poly_gemm_kernel<<<B_ROWS / 32, 256, 0, stream>>>(X, Upk, C, beta, out);
    // 4) finalize scalar regularizer
    reg_final_kernel<<<1, 256, 0, stream>>>(part, C, out + B_ROWS);
}
